// GAT_46523085750849
// MI455X (gfx1250) — matrix operand order repaired and checked by reading, not compiled, not run on hardware
//
#include <hip/hip_runtime.h>
#include <hip/hip_bf16.h>
#include <math.h>

typedef __attribute__((ext_vector_type(16))) _Float16 v16h;
typedef __attribute__((ext_vector_type(8)))  _Float16 v8h;
typedef __attribute__((ext_vector_type(8)))  float    v8f;

#define NEG_SLOPE 0.2f
#define EPSF 1e-16f

// ---------------------------------------------------------------- utilities

__device__ __forceinline__ void atomicMaxFloat(float* addr, float val) {
    // standard signed/unsigned bit trick; works with -inf init
    if (val >= 0.0f) atomicMax((int*)addr, __float_as_int(val));
    else             atomicMin((unsigned int*)addr, (unsigned int)__float_as_int(val));
}

__global__ __launch_bounds__(256) void fill_f32(float* __restrict__ p, float v, long n) {
    long i = (long)blockIdx.x * blockDim.x + threadIdx.x;
    if (i < n) p[i] = v;
}

// gather node embeddings and convert to f16:  h16[n, c] = emb[x[n], c]
__global__ __launch_bounds__(256) void embed_gather(const int* __restrict__ x,
                                                    const float* __restrict__ emb,
                                                    _Float16* __restrict__ h16,
                                                    int N, int D) {
    long i = (long)blockIdx.x * blockDim.x + threadIdx.x;
    if (i >= (long)N * D) return;
    int n = (int)(i / D), c = (int)(i % D);
    h16[i] = (_Float16)emb[(size_t)x[n] * D + c];
}

// W [K, Nout] f32 row-major  ->  Wt [Nout, K] f16 (K contiguous per output col)
__global__ __launch_bounds__(256) void conv_transpose_f16(const float* __restrict__ W,
                                                          _Float16* __restrict__ Wt,
                                                          int K, int Nout) {
    long i = (long)blockIdx.x * blockDim.x + threadIdx.x;
    if (i >= (long)K * Nout) return;
    int k = (int)(i / Nout), n = (int)(i % Nout);
    Wt[(size_t)n * K + k] = (_Float16)W[i];
}

// ------------------------------------------------------------- WMMA GEMM
// C[M,Nout] = A[M,K] (f16 row-major) x Bt[Nout,K] (f16, pre-transposed).
// One wave32 computes a 16 x (16*NT) output strip: A loaded once per k-step,
// reused across NT column tiles -> 4 WMMAs per 10 B128 loads. K % 32 == 0,
// Nout % (16*NT) == 0.
template <bool STORE_HALF, int NT>
__global__ __launch_bounds__(256) void gemm_wmma_f16(const _Float16* __restrict__ A,
                                                     const _Float16* __restrict__ Bt,
                                                     void* __restrict__ Cout,
                                                     int M, int K, int Nout) {
    const int lane = threadIdx.x & 31;
    const int wave = threadIdx.x >> 5;
    const int strip = blockIdx.x * (blockDim.x >> 5) + wave;
    const int Mtiles = (M + 15) >> 4;
    const int Ngroups = Nout / (16 * NT);
    if (strip >= Mtiles * Ngroups) return;
    const int mt = strip / Ngroups, ng = strip % Ngroups;
    const int hi   = lane >> 4;                   // lane half-group 0/1
    const int mrow = (mt << 4) + (lane & 15);
    const int mclamp = mrow < M ? mrow : (M - 1);

    const _Float16* __restrict__ arow = A + (size_t)mclamp * K;
    const _Float16* __restrict__ brow[NT];
    int ncol[NT];
#pragma unroll
    for (int j = 0; j < NT; ++j) {
        ncol[j] = (ng * NT + j) * 16 + (lane & 15);
        brow[j] = Bt + (size_t)ncol[j] * K;
    }

    v8f acc[NT];
#pragma unroll
    for (int j = 0; j < NT; ++j) acc[j] = (v8f){};

    for (int k0 = 0; k0 < K; k0 += 32) {
        // A-matrix lane layout (16-bit A 16x32): halves 0..7 -> K=k0+hi*8+i,
        // halves 8..15 -> K=k0+16+hi*8+i  (two contiguous 16B loads)
        v8h alo = *(const v8h*)(arow + k0 + hi * 8);
        v8h ahi = *(const v8h*)(arow + k0 + 16 + hi * 8);
        v16h a;
#pragma unroll
        for (int i = 0; i < 8; ++i) { a[i] = alo[i]; a[i + 8] = ahi[i]; }
#pragma unroll
        for (int j = 0; j < NT; ++j) {
            // B-matrix: lane n = l&15, halves 0..7 -> K = k0 + hi*8 + i, halves 8..15 -> K = k0 + 16 + hi*8 + i
            v8h blo = *(const v8h*)(brow[j] + k0 + hi * 8);
            v8h bhi = *(const v8h*)(brow[j] + k0 + 16 + hi * 8);
            v16h b;
#pragma unroll
            for (int i = 0; i < 8; ++i) { b[i] = blo[i]; b[i + 8] = bhi[i]; }
            acc[j] = __builtin_amdgcn_wmma_f32_16x16x32_f16(
                false, a, false, b, (short)0, acc[j], false, false);
        }
    }
    // D layout: VGPR r -> row (mt*16 + hi*8 + r), col ncol[j]
#pragma unroll
    for (int j = 0; j < NT; ++j) {
#pragma unroll
        for (int r = 0; r < 8; ++r) {
            int row = (mt << 4) + hi * 8 + r;
            if (row >= M) continue;
            if (STORE_HALF)
                ((_Float16*)Cout)[(size_t)row * Nout + ncol[j]] = (_Float16)acc[j][r];
            else
                ((float*)Cout)[(size_t)row * Nout + ncol[j]] = acc[j][r];
        }
    }
}

// ---------------------------------------------------- attention coefficients
// per (node, head): as[n,h] = dot(h[n,h,:], att_src[h,:]); same for ad.
__global__ __launch_bounds__(256) void node_alpha_h16(const _Float16* __restrict__ h,
                                                      const float* __restrict__ ws,
                                                      const float* __restrict__ wd,
                                                      float* __restrict__ as,
                                                      float* __restrict__ ad,
                                                      int N, int H, int C) {
    int lane = threadIdx.x & 31;
    int w = blockIdx.x * (blockDim.x >> 5) + (threadIdx.x >> 5);
    if (w >= N * H) return;
    int n = w / H, hh = w % H;
    const _Float16* hp = h + (size_t)n * H * C + (size_t)hh * C;
    const float* s = ws + hh * C;
    const float* d = wd + hh * C;
    float ss = 0.f, sd = 0.f;
    for (int c = lane; c < C; c += 32) {
        float v = (float)hp[c];
        ss += v * s[c];
        sd += v * d[c];
    }
    for (int m = 16; m; m >>= 1) { ss += __shfl_xor(ss, m, 32); sd += __shfl_xor(sd, m, 32); }
    if (!lane) { as[w] = ss; ad[w] = sd; }
}

__global__ __launch_bounds__(256) void node_alpha_f32(const float* __restrict__ h,
                                                      const float* __restrict__ ws,
                                                      const float* __restrict__ wd,
                                                      float* __restrict__ as,
                                                      float* __restrict__ ad,
                                                      int N, int H, int C) {
    int lane = threadIdx.x & 31;
    int w = blockIdx.x * (blockDim.x >> 5) + (threadIdx.x >> 5);
    if (w >= N * H) return;
    int n = w / H, hh = w % H;
    const float* hp = h + (size_t)n * H * C + (size_t)hh * C;
    const float* s = ws + hh * C;
    const float* d = wd + hh * C;
    float ss = 0.f, sd = 0.f;
    for (int c = lane; c < C; c += 32) {
        float v = hp[c];
        ss += v * s[c];
        sd += v * d[c];
    }
    for (int m = 16; m; m >>= 1) { ss += __shfl_xor(ss, m, 32); sd += __shfl_xor(sd, m, 32); }
    if (!lane) { as[w] = ss; ad[w] = sd; }
}

// -------------------------------------------------- segment softmax (3 pass)
__device__ __forceinline__ void edge_endpoints(const int* ei, int E, int e, int& src, int& dst) {
    if (e < E) { src = ei[e]; dst = ei[E + e]; }
    else       { src = dst = e - E; }          // self loops appended
}

// pass 1: e = leaky_relu(as[src]+ad[dst]); m[dst] = max(m[dst], e)
__global__ __launch_bounds__(256) void edge_scores(const int* __restrict__ ei, int E, int N, int H,
                                                   const float* __restrict__ as,
                                                   const float* __restrict__ ad,
                                                   float* __restrict__ ebuf,
                                                   float* __restrict__ mbuf) {
    long i = (long)blockIdx.x * blockDim.x + threadIdx.x;
    long tot = (long)(E + N) * H;
    if (i >= tot) return;
    int e = (int)(i / H), hh = (int)(i % H);
    int src, dst; edge_endpoints(ei, E, e, src, dst);
    float v = as[(size_t)src * H + hh] + ad[(size_t)dst * H + hh];
    v = v > 0.f ? v : NEG_SLOPE * v;
    ebuf[i] = v;
    atomicMaxFloat(mbuf + (size_t)dst * H + hh, v);
}

// pass 2: p = exp(e - m[dst]); s[dst] += p  (p overwrites ebuf)
__global__ __launch_bounds__(256) void edge_exp(const int* __restrict__ ei, int E, int N, int H,
                                                float* __restrict__ ebuf,
                                                const float* __restrict__ mbuf,
                                                float* __restrict__ sbuf) {
    long i = (long)blockIdx.x * blockDim.x + threadIdx.x;
    long tot = (long)(E + N) * H;
    if (i >= tot) return;
    int e = (int)(i / H), hh = (int)(i % H);
    int src, dst; edge_endpoints(ei, E, e, src, dst);
    (void)src;
    float p = __expf(ebuf[i] - mbuf[(size_t)dst * H + hh]);
    ebuf[i] = p;
    atomicAdd(sbuf + (size_t)dst * H + hh, p);
}

// pass 3: out[dst] += h[src] * p/(s[dst]+eps)   -- one wave per edge
__global__ __launch_bounds__(256) void edge_aggr_h16(const int* __restrict__ ei, int E, int N,
                                                     int H, int C,
                                                     const _Float16* __restrict__ hsrc,
                                                     const float* __restrict__ p,
                                                     const float* __restrict__ s,
                                                     float* __restrict__ out) {
    int lane = threadIdx.x & 31;
    int e = blockIdx.x * (blockDim.x >> 5) + (threadIdx.x >> 5);
    if (e >= E + N) return;
    int src, dst; edge_endpoints(ei, E, e, src, dst);
    const _Float16* hp = hsrc + (size_t)src * H * C;
    float* op = out + (size_t)dst * H * C;
    for (int hh = 0; hh < H; ++hh) {
        float alpha = p[(size_t)e * H + hh] / (s[(size_t)dst * H + hh] + EPSF);
        for (int c = lane; c < C; c += 32)
            atomicAdd(op + hh * C + c, (float)hp[hh * C + c] * alpha);
    }
}

__global__ __launch_bounds__(256) void edge_aggr_f32(const int* __restrict__ ei, int E, int N,
                                                     int H, int C,
                                                     const float* __restrict__ hsrc,
                                                     const float* __restrict__ p,
                                                     const float* __restrict__ s,
                                                     float* __restrict__ out) {
    int lane = threadIdx.x & 31;
    int e = blockIdx.x * (blockDim.x >> 5) + (threadIdx.x >> 5);
    if (e >= E + N) return;
    int src, dst; edge_endpoints(ei, E, e, src, dst);
    const float* hp = hsrc + (size_t)src * H * C;
    float* op = out + (size_t)dst * H * C;
    for (int hh = 0; hh < H; ++hh) {
        float alpha = p[(size_t)e * H + hh] / (s[(size_t)dst * H + hh] + EPSF);
        for (int c = lane; c < C; c += 32)
            atomicAdd(op + hh * C + c, hp[hh * C + c] * alpha);
    }
}

// ------------------------------------------------------- bias + ELU epilogs
__global__ __launch_bounds__(256) void bias_elu_to_half(const float* __restrict__ in,
                                                        const float* __restrict__ b,
                                                        _Float16* __restrict__ out,
                                                        long total, int C) {
    long i = (long)blockIdx.x * blockDim.x + threadIdx.x;
    if (i >= total) return;
    float v = in[i] + b[i % C];
    v = v > 0.f ? v : (expf(v) - 1.0f);
    out[i] = (_Float16)v;
}

__global__ __launch_bounds__(256) void bias_elu_inplace(float* __restrict__ io,
                                                        const float* __restrict__ b,
                                                        long total, int C) {
    long i = (long)blockIdx.x * blockDim.x + threadIdx.x;
    if (i >= total) return;
    float v = io[i] + b[i % C];
    io[i] = v > 0.f ? v : (expf(v) - 1.0f);
}

// ------------------------------------------------------------------ pooling
__global__ __launch_bounds__(256) void pool_acc(const int* __restrict__ batch,
                                                const float* __restrict__ h,
                                                float* __restrict__ sums,
                                                float* __restrict__ cnt,
                                                int N, int C) {
    long i = (long)blockIdx.x * blockDim.x + threadIdx.x;
    if (i >= (long)N * C) return;
    int n = (int)(i / C), c = (int)(i % C);
    int g = batch[n];
    atomicAdd(&sums[(size_t)g * C + c], h[i]);
    if (c == 0) atomicAdd(&cnt[g], 1.0f);
}

// final: out[g] = dot(pooled[g], Wc) + bc  (one wave per graph)
__global__ __launch_bounds__(256) void final_out(const float* __restrict__ sums,
                                                 const float* __restrict__ cnt,
                                                 const float* __restrict__ Wc,
                                                 const float* __restrict__ bc,
                                                 float* __restrict__ out, int G, int C) {
    int lane = threadIdx.x & 31;
    int g = blockIdx.x * (blockDim.x >> 5) + (threadIdx.x >> 5);
    if (g >= G) return;
    float cn = cnt[g]; cn = cn > 1.f ? cn : 1.f;
    float acc = 0.f;
    for (int c = lane; c < C; c += 32) acc += (sums[(size_t)g * C + c] / cn) * Wc[c];
    for (int m = 16; m; m >>= 1) acc += __shfl_xor(acc, m, 32);
    if (!lane) out[g] = acc + bc[0];
}

// =========================================================== host dispatch

static inline int nblk(long n, int t) { return (int)((n + t - 1) / t); }

extern "C" void kernel_launch(void* const* d_in, const int* in_sizes, int n_in,
                              void* d_out, int out_size, void* d_ws, size_t ws_size,
                              hipStream_t stream) {
    const int*   x    = (const int*)  d_in[0];
    const int*   ei   = (const int*)  d_in[1];   // [2,E] flat
    const int*   batch= (const int*)  d_in[2];
    const float* emb  = (const float*)d_in[3];
    const float* W1   = (const float*)d_in[4];
    const float* as1w = (const float*)d_in[5];
    const float* ad1w = (const float*)d_in[6];
    const float* b1   = (const float*)d_in[7];
    const float* W2   = (const float*)d_in[8];
    const float* as2w = (const float*)d_in[9];
    const float* ad2w = (const float*)d_in[10];
    const float* b2   = (const float*)d_in[11];
    const float* Wc   = (const float*)d_in[12];
    const float* bc   = (const float*)d_in[13];
    float* out = (float*)d_out;

    const int N  = in_sizes[0];
    const int E  = in_sizes[1] / 2;
    const int E2 = E + N;                   // with self loops
    const int D  = 256;                     // embedding dim
    const int H1 = 4, C1 = 128, F1 = H1 * C1;   // layer-1: 4 heads x 128 = 512
    const int C2 = 64;                      // layer-2 out channels (1 head)
    const int G  = 64;                      // graphs

    // ---------------- workspace layout (256B aligned slices) ----------------
    char* base = (char*)d_ws;
    size_t off = 0;
    auto take = [&](size_t bytes) -> char* {
        char* p = base + off;
        off = (off + bytes + 255) & ~(size_t)255;
        return p;
    };
    _Float16* h16    = (_Float16*)take((size_t)N * D  * 2);   // gathered embeddings
    _Float16* w1t    = (_Float16*)take((size_t)F1 * D * 2);   // W1 transposed f16
    _Float16* h1h    = (_Float16*)take((size_t)N * F1 * 2);   // GEMM1 output (pre-act)
    float*    as1    = (float*)   take((size_t)N * H1 * 4);
    float*    ad1    = (float*)   take((size_t)N * H1 * 4);
    float*    m1     = (float*)   take((size_t)N * H1 * 4);
    float*    s1     = (float*)   take((size_t)N * H1 * 4);
    float*    e1     = (float*)   take((size_t)E2 * H1 * 4);  // scores -> probs
    float*    out1   = (float*)   take((size_t)N * F1 * 4);   // aggregated layer-1
    _Float16* hact16 = (_Float16*)take((size_t)N * F1 * 2);   // elu(out1+b1) f16
    _Float16* w2t    = (_Float16*)take((size_t)C2 * F1 * 2);  // W2 transposed f16
    float*    h2     = (float*)   take((size_t)N * C2 * 4);   // GEMM2 output (pre-act)
    float*    as2    = (float*)   take((size_t)N * 4);
    float*    ad2    = (float*)   take((size_t)N * 4);
    float*    m2     = (float*)   take((size_t)N * 4);
    float*    s2     = (float*)   take((size_t)N * 4);
    float*    e2     = (float*)   take((size_t)E2 * 4);
    float*    out2   = (float*)   take((size_t)N * C2 * 4);   // aggregated layer-2
    float*    sums   = (float*)   take((size_t)G * C2 * 4);
    float*    cnt    = (float*)   take((size_t)G * 4);
    (void)ws_size; (void)n_in; (void)out_size;

    const int T = 256;

    // zero accumulators (graph-capturable memset nodes)
    hipMemsetAsync(s1,   0, (size_t)N * H1 * 4, stream);
    hipMemsetAsync(out1, 0, (size_t)N * F1 * 4, stream);
    hipMemsetAsync(s2,   0, (size_t)N * 4,      stream);
    hipMemsetAsync(out2, 0, (size_t)N * C2 * 4, stream);
    hipMemsetAsync(sums, 0, (size_t)G * C2 * 4, stream);
    hipMemsetAsync(cnt,  0, (size_t)G * 4,      stream);
    fill_f32<<<nblk((long)N * H1, T), T, 0, stream>>>(m1, -INFINITY, (long)N * H1);
    fill_f32<<<nblk((long)N, T), T, 0, stream>>>(m2, -INFINITY, (long)N);

    // -------- stage 0: embedding gather (f32 -> f16) + weight prep ----------
    embed_gather<<<nblk((long)N * D, T), T, 0, stream>>>(x, emb, h16, N, D);
    conv_transpose_f16<<<nblk((long)D * F1, T), T, 0, stream>>>(W1, w1t, D, F1);
    conv_transpose_f16<<<nblk((long)F1 * C2, T), T, 0, stream>>>(W2, w2t, F1, C2);

    // -------- stage 1: GEMM1 via WMMA:  h1h = h16 @ W1  (f16 out) -----------
    {
        int strips = ((N + 15) / 16) * (F1 / 64);   // 16x64 strips, NT=4
        gemm_wmma_f16<true, 4><<<nblk(strips, T / 32), T, 0, stream>>>(h16, w1t, h1h, N, D, F1);
    }

    // -------- stage 2: layer-1 attention ------------------------------------
    node_alpha_h16<<<nblk((long)N * H1, T / 32), T, 0, stream>>>(h1h, as1w, ad1w, as1, ad1, N, H1, C1);
    edge_scores<<<nblk((long)E2 * H1, T), T, 0, stream>>>(ei, E, N, H1, as1, ad1, e1, m1);
    edge_exp   <<<nblk((long)E2 * H1, T), T, 0, stream>>>(ei, E, N, H1, e1, m1, s1);
    edge_aggr_h16<<<nblk(E2, T / 32), T, 0, stream>>>(ei, E, N, H1, C1, h1h, e1, s1, out1);
    bias_elu_to_half<<<nblk((long)N * F1, T), T, 0, stream>>>(out1, b1, hact16, (long)N * F1, F1);

    // -------- stage 3: GEMM2 via WMMA:  h2 = hact16 @ W2  (f32 out) ---------
    {
        int strips = ((N + 15) / 16) * (C2 / 64);   // 16x64 strips, NT=4
        gemm_wmma_f16<false, 4><<<nblk(strips, T / 32), T, 0, stream>>>(hact16, w2t, h2, N, F1, C2);
    }

    // -------- stage 4: layer-2 attention (1 head, 64 ch) --------------------
    node_alpha_f32<<<nblk((long)N, T / 32), T, 0, stream>>>(h2, as2w, ad2w, as2, ad2, N, 1, C2);
    edge_scores<<<nblk((long)E2, T), T, 0, stream>>>(ei, E, N, 1, as2, ad2, e2, m2);
    edge_exp   <<<nblk((long)E2, T), T, 0, stream>>>(ei, E, N, 1, e2, m2, s2);
    edge_aggr_f32<<<nblk(E2, T / 32), T, 0, stream>>>(ei, E, N, 1, C2, h2, e2, s2, out2);
    bias_elu_inplace<<<nblk((long)N * C2, T), T, 0, stream>>>(out2, b2, (long)N * C2, C2);

    // -------- stage 5: global mean pool + classifier ------------------------
    pool_acc<<<nblk((long)N * C2, T), T, 0, stream>>>(batch, out2, sums, cnt, N, C2);
    final_out<<<nblk(G, T / 32), T, 0, stream>>>(sums, cnt, Wc, bc, out, G, C2);
}
